// TrackHeadTASUF_14164802142831
// MI455X (gfx1250) — compile-verified, matrix operand order checked by reading, not run on hardware
//
#include <hip/hip_runtime.h>
#include <hip/hip_bf16.h>

typedef __bf16 bf16_t;
typedef __attribute__((ext_vector_type(16))) __bf16 v16bf;
typedef __attribute__((ext_vector_type(8)))  float  v8f;
typedef int v4i_vs __attribute__((vector_size(16)));   // 128-bit chunk type for async-LDS builtin

#define BK 32
#define LP 40   // LDS row pitch in bf16 elements (32 + 8 pad -> 80B rows, keeps 16B alignment)

// ---------------------------------------------------------------- utilities
__global__ __launch_bounds__(256) void zero_u32_kernel(unsigned int* __restrict__ p, long long n) {
    long long i  = (long long)blockIdx.x * blockDim.x + threadIdx.x;
    long long st = (long long)gridDim.x * blockDim.x;
    for (; i < n; i += st) p[i] = 0u;
}

__global__ __launch_bounds__(256) void f32_to_bf16_kernel(const float* __restrict__ in,
                                                          bf16_t* __restrict__ out, long long n) {
    long long i  = (long long)blockIdx.x * blockDim.x + threadIdx.x;
    long long st = (long long)gridDim.x * blockDim.x;
    for (; i < n; i += st) out[i] = (bf16_t)in[i];
}

// 16B global->LDS stage. Prefer the gfx1250 async-to-LDS path (ASYNCcnt-tracked,
// no VGPR round-trip); fall back to load+store staging if the builtin is absent.
// Probe round established: param0 is v4i addrspace(1)*, 4 args total.
__device__ __forceinline__ void stage16(const bf16_t* __restrict__ g, bf16_t* __restrict__ l) {
#if defined(__gfx1250__) && __has_builtin(__builtin_amdgcn_global_load_async_to_lds_b128)
    __builtin_amdgcn_global_load_async_to_lds_b128(
        (__attribute__((address_space(1))) v4i_vs*)(void*)g,
        (__attribute__((address_space(3))) v4i_vs*)(void*)l, 0, 0);
#else
    *(uint4*)l = *(const uint4*)g;
#endif
}

__device__ __forceinline__ void stage_fence() {
#if defined(__gfx1250__) && __has_builtin(__builtin_amdgcn_global_load_async_to_lds_b128)
  #if __has_builtin(__builtin_amdgcn_s_wait_asynccnt)
    __builtin_amdgcn_s_wait_asynccnt(0);
  #else
    asm volatile("s_wait_asynccnt 0x0" ::: "memory");
  #endif
#endif
}

// Fragment gather: per the 16-bit operand layout, one lane's 16 elements are two
// contiguous 8-element (16B) runs of a single tile row selected by the lane half:
//   elems[0..7]  = row[half*8      .. half*8+7 ]   (K 0..15 group)
//   elems[8..15] = row[16 + half*8 .. 16+half*8+7] (K 16..31 group)
// -> exactly two ds_load_b128 per fragment.
__device__ __forceinline__ v16bf load_frag(const bf16_t* __restrict__ row, int kh) {
    union { v16bf v; uint4 q[2]; } f;
    f.q[0] = *(const uint4*)(row + kh * 8);
    f.q[1] = *(const uint4*)(row + 16 + kh * 8);
    return f.v;
}

// ---------------------------------------------------------------- WMMA GEMM
// C[m,n] = sum_k A[m,k]*B[n,k] (+ sum_k A2[m,k]*B2[n,k]) + bias[n] + bias2[n]
// A: [M,K] bf16 row-major, B: [N,K] bf16 row-major ("NT": both contiguous along K).
// Block tile TBM x TBN, 8 waves; wave tile 32 x WN -> 2 A-frags reused over WN/16 B-frags.
template <int TBM, int TBN, int WN>
__global__ __launch_bounds__(256)
void gemm_bf16_wmma(const bf16_t* __restrict__ A,  const bf16_t* __restrict__ B,
                    const bf16_t* __restrict__ A2, const bf16_t* __restrict__ B2,
                    const float* __restrict__ bias, const float* __restrict__ bias2,
                    float* __restrict__ Cf, bf16_t* __restrict__ Cb,
                    int K, int K2, int ldc,
                    long long strideA, long long strideB, long long strideC,
                    int relu)
{
    constexpr int WAVES_N = TBN / WN;   // waves along N
    constexpr int BI      = WN / 16;    // B-subtiles per wave
    static_assert((TBM / 32) * WAVES_N == 8, "8 waves per block");

    __shared__ __align__(16) bf16_t As[TBM][LP];
    __shared__ __align__(16) bf16_t Bs[TBN][LP];

    const int tid  = threadIdx.x;
    const int lane = tid & 31;
    const int wave = tid >> 5;
    const int bM   = blockIdx.x * TBM;
    const int bN   = blockIdx.y * TBN;
    const int z    = blockIdx.z;

    const int wm = (wave / WAVES_N) * 32;
    const int wn = (wave % WAVES_N) * WN;

    A += (long long)z * strideA;
    B += (long long)z * strideB;

    // cooperative tile loaders: 256 threads, 16B chunks, TBM/64 (resp. TBN/64) rows each
    const int lr = tid >> 2;          // 0..63
    const int lc = (tid & 3) * 8;     // 0,8,16,24 bf16 elems (16B chunks)

    v8f acc[2][BI] = {};

    for (int src = 0; src < 2; ++src) {
        const bf16_t* Ap = src ? A2 : A;
        const bf16_t* Bp = src ? B2 : B;
        const int     Kc = src ? K2 : K;
        if (Ap == nullptr) continue;           // wave-uniform

        for (int k0 = 0; k0 < Kc; k0 += BK) {
            __syncthreads();
            #pragma unroll
            for (int i = 0; i < TBM / 64; ++i)
                stage16(Ap + (long long)(bM + i * 64 + lr) * Kc + k0 + lc, &As[i * 64 + lr][lc]);
            #pragma unroll
            for (int i = 0; i < TBN / 64; ++i)
                stage16(Bp + (long long)(bN + i * 64 + lr) * Kc + k0 + lc, &Bs[i * 64 + lr][lc]);
            if (k0 + BK < Kc) {   // pull next k-strip toward the WGP (global_prefetch_b8)
                __builtin_prefetch(Ap + (long long)(bM + lr) * Kc + k0 + BK + lc, 0, 3);
                __builtin_prefetch(Bp + (long long)(bN + lr) * Kc + k0 + BK + lc, 0, 3);
            }
            stage_fence();
            __syncthreads();

            const int kh = lane >> 4;
            const int lm = lane & 15;
            v16bf af[2], bfr[BI];
            #pragma unroll
            for (int ai = 0; ai < 2; ++ai)
                af[ai] = load_frag(&As[wm + ai * 16 + lm][0], kh);
            #pragma unroll
            for (int bi = 0; bi < BI; ++bi)
                bfr[bi] = load_frag(&Bs[wn + bi * 16 + lm][0], kh);

            #pragma unroll
            for (int ai = 0; ai < 2; ++ai)
                #pragma unroll
                for (int bi = 0; bi < BI; ++bi)
                    acc[ai][bi] = __builtin_amdgcn_wmma_f32_16x16x32_bf16(
                        false, af[ai], false, bfr[bi], (short)0, acc[ai][bi], false, false);
        }
    }

    // ---- epilogue: C layout = VGPR e -> row (lane>>4)*8+e, col lane&15 ----
    float*  Cfp = Cf ? Cf + (long long)z * strideC : nullptr;
    bf16_t* Cbp = Cb ? Cb + (long long)z * strideC : nullptr;
    const int lm = lane & 15;
    const int eh = (lane >> 4) << 3;

    #pragma unroll
    for (int ai = 0; ai < 2; ++ai) {
        #pragma unroll
        for (int bi = 0; bi < BI; ++bi) {
            const int col = bN + wn + bi * 16 + lm;
            float badd = 0.f;
            if (bias)  badd += bias[col];
            if (bias2) badd += bias2[col];
            #pragma unroll
            for (int e = 0; e < 8; ++e) {
                float v = acc[ai][bi][e] + badd;
                if (relu) v = fmaxf(v, 0.f);
                const long long ro = (long long)(bM + wm + ai * 16 + eh + e) * ldc;
                if (Cfp) Cfp[ro + col] = v;
                if (Cbp) Cbp[ro + col] = (bf16_t)v;
            }
        }
    }
}

// ---------------------------------------------------------------- LSTM cell
__device__ __forceinline__ float sigm(float x) { return 1.0f / (1.0f + expf(-x)); }

__global__ __launch_bounds__(256)
void lstm_pointwise(const float* __restrict__ gates, float* __restrict__ c,
                    bf16_t* __restrict__ h, int Hd, int rows)
{
    int idx = blockIdx.x * blockDim.x + threadIdx.x;
    if (idx >= rows * Hd) return;
    const int r = idx / Hd;
    const int n = idx - r * Hd;
    const float* g = gates + (long long)r * 4 * Hd;   // PyTorch gate order: i f g o
    const float iv = sigm(g[n]);
    const float fv = sigm(g[Hd + n]);
    const float gv = tanhf(g[2 * Hd + n]);
    const float ov = sigm(g[3 * Hd + n]);
    const float cv = fv * c[idx] + iv * gv;
    c[idx] = cv;
    h[idx] = (bf16_t)(ov * tanhf(cv));
}

// ---------------------------------------------------------------- host-side helpers
template <int TBM, int TBN, int WN>
static void launch_gemm(hipStream_t stream,
                        const bf16_t* A, const bf16_t* Bw, const bf16_t* A2, const bf16_t* B2,
                        const float* bias, const float* bias2, float* Cf, bf16_t* Cb,
                        int M, int N, int K, int K2, int ldc,
                        long long sA, long long sB, long long sC, int batch, int relu) {
    dim3 grid(M / TBM, N / TBN, batch);
    gemm_bf16_wmma<TBM, TBN, WN><<<grid, 256, 0, stream>>>(A, Bw, A2, B2, bias, bias2, Cf, Cb,
                                                           K, K2, ldc, sA, sB, sC, relu);
}

// ---------------------------------------------------------------- launcher
extern "C" void kernel_launch(void* const* d_in, const int* in_sizes, int n_in,
                              void* d_out, int out_size, void* d_ws, size_t ws_size,
                              hipStream_t stream) {
    (void)in_sizes; (void)n_in; (void)out_size; (void)ws_size;

    const int Bsz = 8, NDET = 128, NOBJ = 64, T = 8, F = 12544, H = 1024;
    const int MQ = Bsz * NDET;     // 1024 detection rows
    const int MR = T * Bsz * NOBJ; // 4096 reference rows
    const int NL = Bsz * NOBJ;     // 512 LSTM batch

    const float* x    = (const float*)d_in[0];
    const float* refx = (const float*)d_in[1];
    const float* qW1  = (const float*)d_in[2];  const float* qb1 = (const float*)d_in[3];
    const float* qW2  = (const float*)d_in[4];  const float* qb2 = (const float*)d_in[5];
    const float* iW1  = (const float*)d_in[6];  const float* ib1 = (const float*)d_in[7];
    const float* iW2  = (const float*)d_in[8];  const float* ib2 = (const float*)d_in[9];
    const float* Wih1 = (const float*)d_in[10]; const float* Whh1 = (const float*)d_in[11];
    const float* bih1 = (const float*)d_in[12]; const float* bhh1 = (const float*)d_in[13];
    const float* Wih2 = (const float*)d_in[14]; const float* Whh2 = (const float*)d_in[15];
    const float* bih2 = (const float*)d_in[16]; const float* bhh2 = (const float*)d_in[17];

    // workspace carve-up (256B aligned)
    char* ws = (char*)d_ws;
    size_t off = 0;
    auto take = [&](size_t bytes) -> char* {
        char* p = ws + off; off = (off + bytes + 255) & ~(size_t)255; return p;
    };
    bf16_t* xb    = (bf16_t*)take((size_t)MQ * F * 2);
    bf16_t* rb    = (bf16_t*)take((size_t)MR * F * 2);
    bf16_t* qW1b  = (bf16_t*)take((size_t)H * F * 2);
    bf16_t* qW2b  = (bf16_t*)take((size_t)H * H * 2);
    bf16_t* iW1b  = (bf16_t*)take((size_t)H * F * 2);
    bf16_t* iW2b  = (bf16_t*)take((size_t)H * H * 2);
    bf16_t* Wih1b = (bf16_t*)take((size_t)4 * H * H * 2);
    bf16_t* Whh1b = (bf16_t*)take((size_t)4 * H * H * 2);
    bf16_t* Wih2b = (bf16_t*)take((size_t)4 * H * H * 2);
    bf16_t* Whh2b = (bf16_t*)take((size_t)4 * H * H * 2);
    bf16_t* zq    = (bf16_t*)take((size_t)MQ * H * 2);   // relu(x W1^T + b1)
    bf16_t* qf    = (bf16_t*)take((size_t)MQ * H * 2);   // final q (bf16)
    bf16_t* r1    = (bf16_t*)take((size_t)MR * H * 2);
    bf16_t* r2    = (bf16_t*)take((size_t)MR * H * 2);   // LSTM input sequence
    float*  gates = (float*) take((size_t)NL * 4 * H * 4);
    bf16_t* h1    = (bf16_t*)take((size_t)NL * H * 2);
    bf16_t* h2    = (bf16_t*)take((size_t)NL * H * 2);
    float*  c1    = (float*) take((size_t)NL * H * 4);
    float*  c2    = (float*) take((size_t)NL * H * 4);

    auto cvt = [&](const float* src, bf16_t* dst, long long n) {
        f32_to_bf16_kernel<<<4096, 256, 0, stream>>>(src, dst, n);
    };
    auto zero = [&](void* p, long long words) {
        zero_u32_kernel<<<1024, 256, 0, stream>>>((unsigned int*)p, words);
    };

    // 1) fp32 -> bf16 once (compute-bound problem; conversion traffic is cheap vs WMMA math)
    cvt(x,    xb,    (long long)MQ * F);
    cvt(refx, rb,    (long long)MR * F);
    cvt(qW1,  qW1b,  (long long)H * F);
    cvt(qW2,  qW2b,  (long long)H * H);
    cvt(iW1,  iW1b,  (long long)H * F);
    cvt(iW2,  iW2b,  (long long)H * H);
    cvt(Wih1, Wih1b, (long long)4 * H * H);
    cvt(Whh1, Whh1b, (long long)4 * H * H);
    cvt(Wih2, Wih2b, (long long)4 * H * H);
    cvt(Whh2, Whh2b, (long long)4 * H * H);

    // 2) zero LSTM state + output (dummy column 0 stays zero)
    zero(c1, (long long)NL * H);
    zero(c2, (long long)NL * H);
    zero(h1, (long long)NL * H / 2);
    zero(h2, (long long)NL * H / 2);
    zero(d_out, (long long)Bsz * NDET * (NOBJ + 1));

    // 3) query MLP: zq = relu(x qW1^T + qb1); qf = zq qW2^T + qb2
    launch_gemm<128, 128, 64>(stream, xb, qW1b, nullptr, nullptr, qb1, nullptr, nullptr, zq,
                              MQ, H, F, 0, H, 0, 0, 0, 1, 1);
    launch_gemm<128, 128, 64>(stream, zq, qW2b, nullptr, nullptr, qb2, nullptr, nullptr, qf,
                              MQ, H, H, 0, H, 0, 0, 0, 1, 0);

    // 4) reference MLP over all T*N rows
    launch_gemm<128, 128, 64>(stream, rb, iW1b, nullptr, nullptr, ib1, nullptr, nullptr, r1,
                              MR, H, F, 0, H, 0, 0, 0, 1, 1);
    launch_gemm<128, 128, 64>(stream, r1, iW2b, nullptr, nullptr, ib2, nullptr, nullptr, r2,
                              MR, H, H, 0, H, 0, 0, 0, 1, 0);

    // 5) 2-layer LSTM, T steps; gates = x Wih^T + h Whh^T + bih + bhh (fused dual-source GEMM)
    const int pwBlocks = (NL * H + 255) / 256;
    for (int t = 0; t < T; ++t) {
        launch_gemm<128, 128, 64>(stream, r2 + (size_t)t * NL * H, Wih1b, h1, Whh1b,
                                  bih1, bhh1, gates, nullptr,
                                  NL, 4 * H, H, H, 4 * H, 0, 0, 0, 1, 0);
        lstm_pointwise<<<pwBlocks, 256, 0, stream>>>(gates, c1, h1, H, NL);
        launch_gemm<128, 128, 64>(stream, h1, Wih2b, h2, Whh2b,
                                  bih2, bhh2, gates, nullptr,
                                  NL, 4 * H, H, H, 4 * H, 0, 0, 0, 1, 0);
        lstm_pointwise<<<pwBlocks, 256, 0, stream>>>(gates, c2, h2, H, NL);
    }

    // 6) per-batch prod[b] = q[b] @ h[b]^T into d_out[:, :, 1:], ldc = NOBJ+1 (small-N variant)
    launch_gemm<128, 64, 32>(stream, qf, h2, nullptr, nullptr, nullptr, nullptr,
                             (float*)d_out + 1, nullptr,
                             NDET, NOBJ, H, 0, NOBJ + 1,
                             (long long)NDET * H, (long long)NOBJ * H,
                             (long long)NDET * (NOBJ + 1), Bsz, 0);
}
